// MultiHeadAttention_68822555951703
// MI455X (gfx1250) — compile-verified, matrix operand order checked by reading, not run on hardware
//
#include <hip/hip_runtime.h>
#include <stdint.h>

typedef __bf16 bf16_t;
typedef __attribute__((ext_vector_type(4)))  __bf16 v4bf;
typedef __attribute__((ext_vector_type(8)))  __bf16 v8bf;
typedef __attribute__((ext_vector_type(16))) __bf16 v16bf;
typedef __attribute__((ext_vector_type(8)))  float  v8f;
typedef int v4i __attribute__((__vector_size__(16)));

#define B_ 2
#define S_ 2048
#define D_ 1024
#define H_ 16
#define DH_ 64
#define QK_ 4194304UL      /* elements per Q/K/V part (B*H*S*DH) */
#define WA_ELEMS 3145728UL /* 1024*3072 */
#define WO_ELEMS 1048576UL /* 1024*1024 */

// ---------------- helpers ----------------

__device__ __forceinline__ bf16_t f2bf(float f) { return (bf16_t)f; }

__device__ __forceinline__ v8f wmma_bf16(v16bf a, v16bf b, v8f c) {
  return __builtin_amdgcn_wmma_f32_16x16x32_bf16(false, a, false, b, (short)0, c,
                                                 false, false);
}

__device__ __forceinline__ v8f zero8() {
  v8f z;
#pragma unroll
  for (int i = 0; i < 8; ++i) z[i] = 0.f;
  return z;
}

// One fragment pattern serves A (16xK rows) and N-major B: per lane two
// contiguous 16-byte runs ->  2 x b128 loads + shufflevector.
// lane: base = p + (lane&15)*ld + (lane>>4)*8 ; elems[0..7]=+0, [8..15]=+16
__device__ __forceinline__ v16bf load_frag(const bf16_t* p, int ld) {
  int lane = threadIdx.x & 31;
  const bf16_t* rp = p + (lane & 15) * ld + ((lane >> 4) << 3);
  v8bf lo = *reinterpret_cast<const v8bf*>(rp);
  v8bf hi = *reinterpret_cast<const v8bf*>(rp + 16);
  return __builtin_shufflevector(lo, hi, 0, 1, 2, 3, 4, 5, 6, 7, 8, 9, 10, 11,
                                 12, 13, 14, 15);
}

// B fragment from pre-swizzled W: each lane reads 32 contiguous bytes.
__device__ __forceinline__ v16bf load_frag_sw(const bf16_t* block) {
  int lane = threadIdx.x & 31;
  return *reinterpret_cast<const v16bf*>(block + ((lane >> 4) << 8) +
                                         ((lane & 15) << 4));
}

// async global->LDS copy of 16 bytes (per-lane addresses)
__device__ __forceinline__ void async_b128(const bf16_t* g, bf16_t* l) {
  __builtin_amdgcn_global_load_async_to_lds_b128(
      (__attribute__((address_space(1))) v4i*)(uintptr_t)g,
      (__attribute__((address_space(3))) v4i*)(unsigned)(uintptr_t)l, 0, 0);
}

// ---------------- kernel 0: W swizzle (f32 -> fragment-major bf16) ----------
// dst block for (kc=k/32, nt=n/16): 512 elems laid out [hi][col][e16] so a
// lane's 16 B-fragment elements are contiguous.
__global__ __launch_bounds__(256) void swizzle_w_kernel(
    const float* __restrict__ src, unsigned short* __restrict__ dst_raw, int N) {
  bf16_t* dst = reinterpret_cast<bf16_t*>(dst_raw);
  size_t e = ((size_t)blockIdx.x * 256 + threadIdx.x) * 4;
  const float4 f = *reinterpret_cast<const float4*>(src + e);
  int k = (int)(e / (size_t)N);
  int n = (int)(e % (size_t)N);
  int kc = k >> 5, kk = k & 31;
  int hi = (kk >> 3) & 1;
  int el = (kk & 7) + ((kk >> 4) << 3);
  int nt = n >> 4, col = n & 15;
  size_t base = (((size_t)kc * (N >> 4) + nt) << 9) + (hi << 8) + el;
  dst[base + (size_t)(col + 0) * 16] = f2bf(f.x);
  dst[base + (size_t)(col + 1) * 16] = f2bf(f.y);
  dst[base + (size_t)(col + 2) * 16] = f2bf(f.z);
  dst[base + (size_t)(col + 3) * 16] = f2bf(f.w);
}

// ---------------- kernel 1: QKV GEMM ----------------
// C[4096,3072] = x @ W_attn + b. Q,K -> [B,H,S,DH]; V -> [B,H,DH,S] (transposed
// so the P*V B-fragment in attention is the contiguous N-major pattern).
__global__ __launch_bounds__(256) void qkv_gemm_kernel(
    const float* __restrict__ x, const unsigned short* __restrict__ Wsw_raw,
    const float* __restrict__ bias, unsigned short* __restrict__ qkv_raw) {
  const bf16_t* Wsw = reinterpret_cast<const bf16_t*>(Wsw_raw);
  bf16_t* qkv = reinterpret_cast<bf16_t*>(qkv_raw);
  __shared__ bf16_t As[128 * 32];
  const int m0 = blockIdx.x * 128;
  const int n0 = blockIdx.y * 128;
  const int w = threadIdx.x >> 5;
  const int wm = w & 3, wn = w >> 2;
  const int lane = threadIdx.x & 31;
  const int col_l = lane & 15, hi = lane >> 4;

  v8f acc[2][4];
#pragma unroll
  for (int mi = 0; mi < 2; ++mi)
#pragma unroll
    for (int ni = 0; ni < 4; ++ni) acc[mi][ni] = zero8();

  for (int kk = 0; kk < D_; kk += 32) {
    // stage A tile 128x32 (f32 -> bf16); e == rr*32+cc so As index == e
    float4 fa[4];
#pragma unroll
    for (int it = 0; it < 4; ++it) {
      int e = (threadIdx.x + it * 256) * 4;
      fa[it] = *reinterpret_cast<const float4*>(
          x + (size_t)(m0 + (e >> 5)) * D_ + kk + (e & 31));
    }
#pragma unroll
    for (int it = 0; it < 4; ++it) {
      int e = (threadIdx.x + it * 256) * 4;
      v4bf t = {f2bf(fa[it].x), f2bf(fa[it].y), f2bf(fa[it].z), f2bf(fa[it].w)};
      *reinterpret_cast<v4bf*>(&As[e]) = t;
    }
    __syncthreads();
    v16bf af[2];
#pragma unroll
    for (int mi = 0; mi < 2; ++mi)
      af[mi] = load_frag(&As[(wm * 32 + mi * 16) * 32], 32);
#pragma unroll
    for (int ni = 0; ni < 4; ++ni) {
      v16bf bw = load_frag_sw(
          Wsw + (((size_t)(kk >> 5) * 192 + (n0 >> 4) + wn * 4 + ni) << 9));
#pragma unroll
      for (int mi = 0; mi < 2; ++mi)
        acc[mi][ni] = wmma_bf16(af[mi], bw, acc[mi][ni]);
    }
    __syncthreads();
  }

  // bias + scatter (part is uniform per block: N-tile never crosses 1024)
  const int part = (n0 + wn * 64) >> 10;
#pragma unroll
  for (int mi = 0; mi < 2; ++mi)
#pragma unroll
    for (int ni = 0; ni < 4; ++ni)
#pragma unroll
      for (int r = 0; r < 8; ++r) {
        int gr = m0 + wm * 32 + mi * 16 + hi * 8 + r;
        int gc = n0 + wn * 64 + ni * 16 + col_l;
        float v = acc[mi][ni][r] + bias[gc];
        int rem = gc & 1023;
        int h = rem >> 6, dh = rem & 63;
        int b = gr >> 11, s = gr & 2047;
        if (part == 2)  // V transposed [B,H,DH,S]
          qkv[2 * QK_ + ((size_t)(b * H_ + h) * DH_ + dh) * S_ + s] = f2bf(v);
        else
          qkv[(size_t)part * QK_ + (((size_t)(b * H_ + h) * S_ + s) << 6) + dh] =
              f2bf(v);
      }
}

// ---------------- kernel 2: flash attention ----------------
// One block per (b*H+h, 64-query tile); 4 waves x 16 query rows.
// K/V tiles double-buffered via async-to-LDS; partial s_wait_asynccnt(16)
// overlaps next-tile fetch with current-tile WMMA + softmax.
__global__ __launch_bounds__(128) void attn_kernel(
    const unsigned short* __restrict__ qkv_raw,
    const unsigned char* __restrict__ mask,
    unsigned short* __restrict__ attn_raw) {
  const bf16_t* qkv = reinterpret_cast<const bf16_t*>(qkv_raw);
  bf16_t* attn = reinterpret_cast<bf16_t*>(attn_raw);
  __shared__ bf16_t Kt[2][128 * DH_];  // [key][dh]
  __shared__ bf16_t Vt[2][DH_ * 128];  // [dh][key]  (V stored transposed)
  __shared__ bf16_t Pt[4][16 * 128];

  const int bh = blockIdx.x;
  const int b = bh >> 4, h = bh & 15;
  const int q0 = blockIdx.y * 64;
  const int w = threadIdx.x >> 5;
  const int lane = threadIdx.x & 31;
  const int col_l = lane & 15, hi = lane >> 4;

  const bf16_t* Qb = qkv + ((size_t)(b * H_ + h) * S_) * DH_;
  const bf16_t* Kb = Qb + QK_;
  const bf16_t* Vb = qkv + 2 * QK_ + ((size_t)(b * H_ + h) * DH_) * S_;
  const unsigned char* mb = mask + (size_t)b * S_ * S_;

  const bf16_t* qp = Qb + (size_t)(q0 + w * 16) * DH_;
  v16bf aq0 = load_frag(qp, DH_);
  v16bf aq1 = load_frag(qp + 32, DH_);

  // issue 16 async b128 copies per thread for one 128-key K/V tile pair
  auto stage_tile = [&](int kt, int buf) {
#pragma unroll
    for (int it = 0; it < 8; ++it) {
      int i = threadIdx.x + it * 128;  // 0..1023 chunks of 8 bf16
      async_b128(Kb + (size_t)kt * DH_ + i * 8, &Kt[buf][i * 8]);
    }
#pragma unroll
    for (int it = 0; it < 8; ++it) {
      int i = threadIdx.x + it * 128;
      int row = i >> 4, off = (i & 15) * 8;
      async_b128(Vb + (size_t)row * S_ + kt + off, &Vt[buf][row * 128 + off]);
    }
  };

  float m_r[8], l_r[8];
#pragma unroll
  for (int r = 0; r < 8; ++r) { m_r[r] = -3.0e38f; l_r[r] = 0.f; }
  v8f o[4];
#pragma unroll
  for (int ni = 0; ni < 4; ++ni) o[ni] = zero8();

  stage_tile(0, 0);  // prologue: tile 0 into buffer 0

  for (int t = 0; t < 16; ++t) {
    const int cur = t & 1;
    const int kt = t * 128;
    // prefetch next tile (index wraps on last iter -> wait imm stays constant;
    // the stray copy targets the unread buffer and s_endpgm waits it out)
    stage_tile(((t + 1) & 15) * 128, cur ^ 1);
    __builtin_amdgcn_s_wait_asynccnt(16);  // this tile's 16 copies landed
    __syncthreads();

    // S = Q K^T
    v8f sacc[8];
#pragma unroll
    for (int nt = 0; nt < 8; ++nt) {
      v16bf bk0 = load_frag(&Kt[cur][(nt * 16) * DH_], DH_);
      v16bf bk1 = load_frag(&Kt[cur][(nt * 16) * DH_ + 32], DH_);
      v8f z = zero8();
      z = wmma_bf16(aq0, bk0, z);
      sacc[nt] = wmma_bf16(aq1, bk1, z);
    }

    // scale + mask
#pragma unroll
    for (int nt = 0; nt < 8; ++nt)
#pragma unroll
      for (int r = 0; r < 8; ++r) {
        int qg = q0 + w * 16 + hi * 8 + r;
        int kg = kt + nt * 16 + col_l;
        float sv = sacc[nt][r] * 0.125f;
        if (mb[(size_t)qg * S_ + kg]) sv = -1.0e9f;
        sacc[nt][r] = sv;
      }

    // online softmax (row = hi*8+r lives in one 16-lane group)
#pragma unroll
    for (int r = 0; r < 8; ++r) {
      float vmax = -3.0e38f;
#pragma unroll
      for (int nt = 0; nt < 8; ++nt) vmax = fmaxf(vmax, sacc[nt][r]);
#pragma unroll
      for (int off = 1; off < 16; off <<= 1)
        vmax = fmaxf(vmax, __shfl_xor(vmax, off, 32));
      float mnew = fmaxf(m_r[r], vmax);
      float alpha = __expf(m_r[r] - mnew);
      float rsum = 0.f;
#pragma unroll
      for (int nt = 0; nt < 8; ++nt) {
        float p = __expf(sacc[nt][r] - mnew);
        Pt[w][(hi * 8 + r) * 128 + nt * 16 + col_l] = f2bf(p);
        rsum += p;
      }
#pragma unroll
      for (int off = 1; off < 16; off <<= 1) rsum += __shfl_xor(rsum, off, 32);
      l_r[r] = l_r[r] * alpha + rsum;
      m_r[r] = mnew;
#pragma unroll
      for (int ni = 0; ni < 4; ++ni) o[ni][r] *= alpha;
    }

    // O += P V   (V N-major in LDS -> contiguous fragment loads)
    v16bf ap[4];
#pragma unroll
    for (int kc = 0; kc < 4; ++kc) ap[kc] = load_frag(&Pt[w][kc * 32], 128);
#pragma unroll
    for (int ni = 0; ni < 4; ++ni)
#pragma unroll
      for (int kc = 0; kc < 4; ++kc) {
        v16bf bv = load_frag(&Vt[cur][(ni * 16) * 128 + kc * 32], 128);
        o[ni] = wmma_bf16(ap[kc], bv, o[ni]);
      }
    __syncthreads();  // all waves done with buf[cur] before it's restaged
  }

  // normalize + write attn out [B,S,H*DH] bf16
#pragma unroll
  for (int r = 0; r < 8; ++r) {
    float inv = 1.f / l_r[r];
#pragma unroll
    for (int ni = 0; ni < 4; ++ni) {
      int sg = q0 + w * 16 + hi * 8 + r;
      int c = h * DH_ + ni * 16 + col_l;
      attn[((size_t)(b * S_ + sg)) * D_ + c] = f2bf(o[ni][r] * inv);
    }
  }
}

// ---------------- kernel 3: out-proj + residual + LayerNorm ----------------
__global__ __launch_bounds__(256) void proj_ln_kernel(
    const unsigned short* __restrict__ attn_raw, const float* __restrict__ x,
    const unsigned short* __restrict__ Wosw_raw, const float* __restrict__ bo,
    const float* __restrict__ ln_g, const float* __restrict__ ln_b,
    float* __restrict__ out) {
  const bf16_t* attnp = reinterpret_cast<const bf16_t*>(attn_raw);
  const bf16_t* Wosw = reinterpret_cast<const bf16_t*>(Wosw_raw);
  __shared__ bf16_t As[16 * 32];
  __shared__ float red_s[16], red_q[16];
  const int m0 = blockIdx.x * 16;
  const int w = threadIdx.x >> 5;
  const int lane = threadIdx.x & 31;
  const int col_l = lane & 15, hi = lane >> 4;

  v8f acc[8];
#pragma unroll
  for (int ni = 0; ni < 8; ++ni) acc[ni] = zero8();

  for (int kk = 0; kk < D_; kk += 32) {
    {  // stage A tile 16x32 bf16 (2 elems/thread, 4-byte moves)
      int i2 = threadIdx.x * 2;
      *reinterpret_cast<uint32_t*>(&As[i2]) = *reinterpret_cast<const uint32_t*>(
          &attnp[(size_t)(m0 + (i2 >> 5)) * D_ + kk + (i2 & 31)]);
    }
    __syncthreads();
    v16bf a = load_frag(As, 32);
#pragma unroll
    for (int ni = 0; ni < 8; ++ni) {
      v16bf bw =
          load_frag_sw(Wosw + (((size_t)(kk >> 5) * 64 + w * 8 + ni) << 9));
      acc[ni] = wmma_bf16(a, bw, acc[ni]);
    }
    __syncthreads();
  }

  // y = acc + x + b_out ; LN row statistics
  float y[8][8];
  float psum[8], psq[8];
#pragma unroll
  for (int r = 0; r < 8; ++r) { psum[r] = 0.f; psq[r] = 0.f; }
#pragma unroll
  for (int ni = 0; ni < 8; ++ni)
#pragma unroll
    for (int r = 0; r < 8; ++r) {
      int g = m0 + hi * 8 + r;
      int c = w * 128 + ni * 16 + col_l;
      float v = acc[ni][r] + x[(size_t)g * D_ + c] + bo[c];
      y[ni][r] = v;
      psum[r] += v;
      psq[r] += v * v;
    }
#pragma unroll
  for (int r = 0; r < 8; ++r) {
#pragma unroll
    for (int off = 1; off < 16; off <<= 1) {
      psum[r] += __shfl_xor(psum[r], off, 32);
      psq[r] += __shfl_xor(psq[r], off, 32);
    }
  }
  if (threadIdx.x < 16) { red_s[threadIdx.x] = 0.f; red_q[threadIdx.x] = 0.f; }
  __syncthreads();
  if (col_l == 0) {
#pragma unroll
    for (int r = 0; r < 8; ++r) {
      atomicAdd(&red_s[hi * 8 + r], psum[r]);
      atomicAdd(&red_q[hi * 8 + r], psq[r]);
    }
  }
  __syncthreads();
#pragma unroll
  for (int ni = 0; ni < 8; ++ni)
#pragma unroll
    for (int r = 0; r < 8; ++r) {
      int row = hi * 8 + r;
      float mu = red_s[row] * (1.0f / 1024.0f);
      float var = red_q[row] * (1.0f / 1024.0f) - mu * mu;
      float rstd = rsqrtf(var + 1e-5f);
      int g = m0 + row;
      int c = w * 128 + ni * 16 + col_l;
      out[(size_t)g * D_ + c] = (y[ni][r] - mu) * rstd * ln_g[c] + ln_b[c];
    }
}

// ---------------- launcher ----------------
extern "C" void kernel_launch(void* const* d_in, const int* in_sizes, int n_in,
                              void* d_out, int out_size, void* d_ws,
                              size_t ws_size, hipStream_t stream) {
  const float* x = (const float*)d_in[0];
  const float* W_attn = (const float*)d_in[1];
  const float* b_attn = (const float*)d_in[2];
  const float* W_out = (const float*)d_in[3];
  const float* b_out = (const float*)d_in[4];
  const float* ln_g = (const float*)d_in[5];
  const float* ln_b = (const float*)d_in[6];
  const unsigned char* mask = (const unsigned char*)d_in[7];

  unsigned short* qkv = (unsigned short*)d_ws;         // 3*4M bf16
  unsigned short* attn = qkv + 3 * QK_;                // 4M bf16
  unsigned short* wa_sw = attn + QK_;                  // 3M bf16 (swizzled W_attn)
  unsigned short* wo_sw = wa_sw + WA_ELEMS;            // 1M bf16 (swizzled W_out)
  float* out = (float*)d_out;

  swizzle_w_kernel<<<(int)(WA_ELEMS / 1024), 256, 0, stream>>>(W_attn, wa_sw, 3072);
  swizzle_w_kernel<<<(int)(WO_ELEMS / 1024), 256, 0, stream>>>(W_out, wo_sw, 1024);
  qkv_gemm_kernel<<<dim3(32, 24), 256, 0, stream>>>(x, wa_sw, b_attn, qkv);
  attn_kernel<<<dim3(32, 32), 128, 0, stream>>>(qkv, mask, attn);
  proj_ln_kernel<<<256, 256, 0, stream>>>(attn, x, wo_sw, b_out, ln_g, ln_b, out);
}
